// TrainableVarRegBlock_16011638079635
// MI455X (gfx1250) — matrix operand order repaired and checked by reading, not compiled, not run on hardware
//
#include <hip/hip_runtime.h>

typedef __attribute__((ext_vector_type(16))) _Float16 v16h;
typedef __attribute__((ext_vector_type(8)))  _Float16 v8h;
typedef __attribute__((ext_vector_type(8)))  float    v8f;

#define N64 262144
#define N32 32768

__device__ __forceinline__ float mish_f(float x) {
  float sp = (x > 20.f) ? x : log1pf(expf(x));
  return x * tanhf(sp);
}

// ---------------- generic / demons kernels ----------------

__global__ void vr_fill_kernel(float* p, float v, int n) {
  int i = blockIdx.x * blockDim.x + threadIdx.x;
  if (i < n) p[i] = v;
}

__global__ void vr_copy_kernel(const float* __restrict__ s, float* __restrict__ d, int n) {
  int i = blockIdx.x * blockDim.x + threadIdx.x;
  if (i < n) d[i] = s[i];
}

// trilinear downsample x0.5 (exact 8-avg at half-pixel centers) + nearest for mask
__global__ void vr_downsample_kernel(const float* __restrict__ img, const float* __restrict__ mov,
                                     const float* __restrict__ msk,
                                     float* img32, float* mov32, float* msk32) {
  int i = blockIdx.x * blockDim.x + threadIdx.x;
  if (i >= N32) return;
  int x = i & 31, y = (i >> 5) & 31, z = i >> 10;
  int b = ((2 * z) * 64 + 2 * y) * 64 + 2 * x;
  const int o[8] = {0, 1, 64, 65, 4096, 4097, 4160, 4161};
  float si = 0.f, sm = 0.f;
#pragma unroll
  for (int k = 0; k < 8; k++) { si += img[b + o[k]]; sm += mov[b + o[k]]; }
  img32[i] = si * 0.125f;
  mov32[i] = sm * 0.125f;
  msk32[i] = msk[((2 * z + 1) * 64 + 2 * y + 1) * 64 + 2 * x + 1];
}

// trilinear warp with zeros outside (map_coordinates order=1, constant)
__global__ void vr_warp_kernel(const float* __restrict__ img, const float* __restrict__ flow,
                               float* __restrict__ out, int D, int H, int W) {
  int Nv = D * H * W;
  int v = blockIdx.x * blockDim.x + threadIdx.x;
  if (v >= Nv) return;
  int x = v % W, y = (v / W) % H, z = v / (W * H);
  float fz = z + flow[v], fy = y + flow[Nv + v], fx = x + flow[2 * Nv + v];
  float zf = floorf(fz), yf = floorf(fy), xf = floorf(fx);
  int z0 = (int)zf, y0 = (int)yf, x0 = (int)xf;
  float tz = fz - zf, ty = fy - yf, tx = fx - xf;
  float acc = 0.f;
#pragma unroll
  for (int dz = 0; dz < 2; dz++)
#pragma unroll
    for (int dy = 0; dy < 2; dy++)
#pragma unroll
      for (int dx = 0; dx < 2; dx++) {
        int zc = z0 + dz, yc = y0 + dy, xc = x0 + dx;
        if (zc >= 0 && zc < D && yc >= 0 && yc < H && xc >= 0 && xc < W) {
          float wgt = (dz ? tz : 1.f - tz) * (dy ? ty : 1.f - ty) * (dx ? tx : 1.f - tx);
          acc += wgt * img[((size_t)zc * H + yc) * W + xc];
        }
      }
  out[v] = acc;
}

// demon forces (active) + vf update, TAU=1, eps=1e-6
__global__ void vr_demons_update_kernel(const float* __restrict__ wp, const float* __restrict__ fx,
                                        const float* __restrict__ mk, float* __restrict__ vf,
                                        int D, int H, int W) {
  int Nv = D * H * W;
  int v = blockIdx.x * blockDim.x + threadIdx.x;
  if (v >= Nv) return;
  int x = v % W, y = (v / W) % H, z = v / (W * H);
  int sz = H * W, sy = W;
  float c = wp[v];
  float gz = (z == 0) ? wp[v + sz] - c : (z == D - 1) ? c - wp[v - sz] : 0.5f * (wp[v + sz] - wp[v - sz]);
  float gy = (y == 0) ? wp[v + sy] - c : (y == H - 1) ? c - wp[v - sy] : 0.5f * (wp[v + sy] - wp[v - sy]);
  float gx = (x == 0) ? wp[v + 1] - c : (x == W - 1) ? c - wp[v - 1] : 0.5f * (wp[v + 1] - wp[v - 1]);
  float d = fx[v] - c;
  float den = gz * gz + gy * gy + gx * gx + d * d + 1e-6f;
  float s = d / den * mk[v];
  vf[v] += s * gz;
  vf[Nv + v] += s * gy;
  vf[2 * Nv + v] += s * gx;
}

// separable gaussian pass (sigma=1, radius=2), zero padding, 3 channels
__global__ void vr_smooth_axis_kernel(const float* __restrict__ src, float* __restrict__ dst,
                                      int D, int H, int W, int axis) {
  int Nv = D * H * W;
  int i = blockIdx.x * blockDim.x + threadIdx.x;
  if (i >= 3 * Nv) return;
  int v = i % Nv;
  int x = v % W, y = (v / W) % H, z = v / (W * H);
  int p, L, s;
  if (axis == 0) { p = z; L = D; s = H * W; }
  else if (axis == 1) { p = y; L = H; s = W; }
  else { p = x; L = W; s = 1; }
  const float w[5] = {0.0544886845f, 0.2442013420f, 0.4026199469f, 0.2442013420f, 0.0544886845f};
  float acc = 0.f;
#pragma unroll
  for (int t = -2; t <= 2; t++) {
    int q = p + t;
    if (q >= 0 && q < L) acc += w[t + 2] * src[i + t * s];
  }
  dst[i] = acc;
}

// trilinear upsample 32^3 -> 64^3 of 3-channel vf, times 2 (match_vf scale)
__global__ void vr_upsample_vf_kernel(const float* __restrict__ vf32, float* __restrict__ vf64) {
  int i = blockIdx.x * blockDim.x + threadIdx.x;
  if (i >= 3 * N64) return;
  int c = i / N64, v = i % N64;
  int x = v & 63, y = (v >> 6) & 63, z = v >> 12;
  float sz = fminf(fmaxf(z * 0.5f - 0.25f, 0.f), 31.f);
  float sy = fminf(fmaxf(y * 0.5f - 0.25f, 0.f), 31.f);
  float sx = fminf(fmaxf(x * 0.5f - 0.25f, 0.f), 31.f);
  int z0 = (int)sz, y0 = (int)sy, x0 = (int)sx;
  int z1 = min(z0 + 1, 31), y1 = min(y0 + 1, 31), x1 = min(x0 + 1, 31);
  float tz = sz - z0, ty = sy - y0, tx = sx - x0;
  const float* b = vf32 + (size_t)c * N32;
  auto at = [&](int zz, int yy, int xx) { return b[((zz << 5) + yy) * 32 + xx]; };
  float v00 = at(z0, y0, x0) * (1.f - tx) + at(z0, y0, x1) * tx;
  float v01 = at(z0, y1, x0) * (1.f - tx) + at(z0, y1, x1) * tx;
  float v10 = at(z1, y0, x0) * (1.f - tx) + at(z1, y0, x1) * tx;
  float v11 = at(z1, y1, x0) * (1.f - tx) + at(z1, y1, x1) * tx;
  float r = (v00 * (1.f - ty) + v01 * ty) * (1.f - tz) + (v10 * (1.f - ty) + v11 * ty) * tz;
  vf64[i] = r * 2.0f;
}

// stacked input [vf(3), image*mask, warped] -> f16 act layout [voxel][32]
__global__ void vr_build_act0_kernel(const float* __restrict__ vf, const float* __restrict__ image,
                                     const float* __restrict__ msk, const float* __restrict__ warped,
                                     _Float16* __restrict__ act) {
  int i = blockIdx.x * blockDim.x + threadIdx.x;
  if (i >= N64 * 32) return;
  int c = i & 31, v = i >> 5;
  float r = 0.f;
  if (c == 0) r = vf[v];
  else if (c == 1) r = vf[N64 + v];
  else if (c == 2) r = vf[2 * N64 + v];
  else if (c == 3) r = image[v] * msk[v];
  else if (c == 4) r = warped[v];
  act[i] = (_Float16)r;
}

// ---------------- WMMA implicit-GEMM conv3d (3x3x3, SAME) ----------------

// weights (cout,cin,3,3,3) f32 -> fragment-ready f16:
// wf[(((ntile*27+tap)*KB+kb)*32+lane)*16+i] = W[n=ntile*16+lane%16][k=kb*32+(lane/16)*16+i][tap]
__global__ void vr_wprep_kernel(const float* __restrict__ w, _Float16* __restrict__ wf,
                                int cout, int cin, int Ntiles, int KB) {
  int idx = blockIdx.x * blockDim.x + threadIdx.x;
  int total = Ntiles * 27 * KB * 512;
  if (idx >= total) return;
  int i = idx & 15, lane = (idx >> 4) & 31, rest = idx >> 9;
  int kb = rest % KB; rest /= KB;
  int tap = rest % 27;
  int ntile = rest / 27;
  int n = ntile * 16 + (lane & 15);
  int k = kb * 32 + (lane >> 4) * 16 + i;
  float val = (n < cout && k < cin) ? w[((size_t)n * cin + k) * 27 + tap] : 0.f;
  wf[idx] = (_Float16)val;
}

__device__ __forceinline__ v16h vr_load_a(const _Float16* __restrict__ abase, int xx, bool zyok,
                                          int Kpad, int koff, int hi) {
  v8h lo = {}, hp = {};
  if (zyok && (unsigned)xx < 64u) {
    const _Float16* p = abase + (size_t)xx * Kpad + koff + hi * 16;
    lo = *(const v8h*)(p);        // K = koff + hi*16 + 0..7
    hp = *(const v8h*)(p + 8);    // K = koff + hi*16 + 8..15
  }
  v16h a;
#pragma unroll
  for (int i = 0; i < 8; i++) { a[i] = lo[i]; a[8 + i] = hp[i]; }
  return a;
}

// One wave owns a full x-row (64 voxels) x 16 output channels: 4 independent
// 16x16 accumulators per wave (ILP across WMMAs, B fragment reused 4x).
// B fragments for the block's ntile are staged once into LDS.
__global__ __launch_bounds__(256) void vr_conv3d_wmma_kernel(
    const _Float16* __restrict__ act, const _Float16* __restrict__ wfrag,
    float* __restrict__ out, int Kpad, int Npad, int KB) {
  extern __shared__ _Float16 sB[];  // 27*KB*512 halves
  const int lane = threadIdx.x & 31;
  const int wv = threadIdx.x >> 5;
  const int ntile = blockIdx.y;

  // cooperative stage of this ntile's weight fragments into LDS
  const _Float16* wfb = wfrag + (size_t)ntile * 27 * KB * 512;
  const int nfrag = 27 * KB * 512;
  for (int i = threadIdx.x * 8; i < nfrag; i += 256 * 8)
    *(v8h*)(sB + i) = *(const v8h*)(wfb + i);
  __syncthreads();

  const int mrow = blockIdx.x * 8 + wv;  // 4096 (z,y) rows over 64^3
  const int y = mrow & 63;
  const int z = mrow >> 6;
  const int row = lane & 15, hi = lane >> 4;
  v8f acc[4] = {{}, {}, {}, {}};

  for (int tap = 0; tap < 27; ++tap) {
    const int dz = tap / 9 - 1, dy = (tap / 3) % 3 - 1, dx = tap % 3 - 1;
    const int zz = z + dz, yy = y + dy;
    const bool zyok = ((unsigned)zz < 64u) && ((unsigned)yy < 64u);
    const _Float16* abase = act + (size_t)((zz * 64 + yy) * 64) * Kpad;
    for (int kb = 0; kb < KB; ++kb) {
      v16h b = *(const v16h*)(&sB[(tap * KB + kb) * 512 + lane * 16]);
      v16h a0 = vr_load_a(abase, row + 0 * 16 + dx, zyok, Kpad, kb * 32, hi);
      v16h a1 = vr_load_a(abase, row + 1 * 16 + dx, zyok, Kpad, kb * 32, hi);
      v16h a2 = vr_load_a(abase, row + 2 * 16 + dx, zyok, Kpad, kb * 32, hi);
      v16h a3 = vr_load_a(abase, row + 3 * 16 + dx, zyok, Kpad, kb * 32, hi);
      acc[0] = __builtin_amdgcn_wmma_f32_16x16x32_f16(false, a0, false, b, (short)0, acc[0], false, false);
      acc[1] = __builtin_amdgcn_wmma_f32_16x16x32_f16(false, a1, false, b, (short)0, acc[1], false, false);
      acc[2] = __builtin_amdgcn_wmma_f32_16x16x32_f16(false, a2, false, b, (short)0, acc[2], false, false);
      acc[3] = __builtin_amdgcn_wmma_f32_16x16x32_f16(false, a3, false, b, (short)0, acc[3], false, false);
    }
  }

  const int col = lane & 15;
  const size_t vrow = (size_t)(z * 64 + y) * 64;
#pragma unroll
  for (int sub = 0; sub < 4; ++sub) {
    float* orow = out + (vrow + sub * 16 + hi * 8) * Npad + ntile * 16 + col;
#pragma unroll
    for (int r = 0; r < 8; r++) orow[(size_t)r * Npad] = acc[sub][r];
  }
}

// per-channel sum / sumsq via LDS float atomics + global atomics; sums[0..63]=sum, sums[64..127]=sumsq
__global__ __launch_bounds__(256) void vr_inorm_reduce_kernel(const float* __restrict__ xbuf,
                                                              float* __restrict__ sums,
                                                              int Npad, int Nvox) {
  __shared__ float ssum[64], ssq[64];
  for (int i = threadIdx.x; i < Npad; i += blockDim.x) { ssum[i] = 0.f; ssq[i] = 0.f; }
  __syncthreads();
  int c = threadIdx.x % Npad;
  int vper = blockDim.x / Npad;
  int vstart = blockIdx.x * vper + threadIdx.x / Npad;
  int vstride = gridDim.x * vper;
  float ls = 0.f, lq = 0.f;
  for (int v = vstart; v < Nvox; v += vstride) {
    float val = xbuf[(size_t)v * Npad + c];
    ls += val; lq += val * val;
  }
  atomicAdd(&ssum[c], ls);
  atomicAdd(&ssq[c], lq);
  __syncthreads();
  for (int i = threadIdx.x; i < Npad; i += blockDim.x) {
    atomicAdd(&sums[i], ssum[i]);
    atomicAdd(&sums[64 + i], ssq[i]);
  }
}

// inorm + mish, pack to f16 act layout [voxel][KpadNext] with zero channel padding
__global__ void vr_inorm_mish_pack_kernel(const float* __restrict__ xbuf, const float* __restrict__ sums,
                                          _Float16* __restrict__ actout,
                                          int Npad, int cout, int KpadNext, int Nvox) {
  int idx = blockIdx.x * blockDim.x + threadIdx.x;
  if (idx >= Nvox * KpadNext) return;
  int c = idx % KpadNext, v = idx / KpadNext;
  float r = 0.f;
  if (c < cout) {
    float invN = 1.0f / (float)Nvox;
    float mean = sums[c] * invN;
    float var = sums[64 + c] * invN - mean * mean;
    float val = (xbuf[(size_t)v * Npad + c] - mean) * rsqrtf(var + 1e-5f);
    r = mish_f(val);
  }
  actout[idx] = (_Float16)r;
}

// vf + correction + bias -> raw corrected vf (channel-major)
__global__ void vr_add_corr_kernel(const float* __restrict__ vf, const float* __restrict__ conv4,
                                   const float* __restrict__ b4, float* __restrict__ out) {
  int i = blockIdx.x * blockDim.x + threadIdx.x;
  if (i >= 3 * N64) return;
  int c = i / N64, v = i % N64;
  out[i] = vf[i] + conv4[(size_t)v * 16 + c] + b4[c];
}

// ---------------- host orchestration ----------------

extern "C" void kernel_launch(void* const* d_in, const int* in_sizes, int n_in,
                              void* d_out, int out_size, void* d_ws, size_t ws_size,
                              hipStream_t stream) {
  (void)in_sizes; (void)n_in; (void)out_size; (void)ws_size;
  const float* image  = (const float*)d_in[0];
  const float* mask   = (const float*)d_in[1];
  const float* moving = (const float*)d_in[2];
  const float* w1 = (const float*)d_in[3];
  const float* w2 = (const float*)d_in[4];
  const float* w3 = (const float*)d_in[5];
  const float* w4 = (const float*)d_in[6];
  const float* b4 = (const float*)d_in[7];
  float* out = (float*)d_out;

  char* ws = (char*)d_ws;
  size_t off = 0;
  auto alloc = [&](size_t bytes) -> char* {
    char* p = ws + off;
    off = (off + bytes + 255) & ~(size_t)255;
    return p;
  };
  float*    img32   = (float*)alloc((size_t)N32 * 4);
  float*    mov32   = (float*)alloc((size_t)N32 * 4);
  float*    msk32   = (float*)alloc((size_t)N32 * 4);
  float*    vf32    = (float*)alloc((size_t)3 * N32 * 4);
  float*    vf64    = (float*)alloc((size_t)3 * N64 * 4);
  float*    tmpA    = (float*)alloc((size_t)3 * N64 * 4);
  float*    tmpB    = (float*)alloc((size_t)3 * N64 * 4);
  float*    cvf     = (float*)alloc((size_t)3 * N64 * 4);
  float*    warped  = (float*)alloc((size_t)N64 * 4);
  float*    sums    = (float*)alloc(512);
  _Float16* wfrag   = (_Float16*)alloc((size_t)4 * 27 * 2 * 512 * 2);
  _Float16* actA    = (_Float16*)alloc((size_t)N64 * 64 * 2);
  _Float16* actB    = (_Float16*)alloc((size_t)N64 * 64 * 2);
  float*    convout = (float*)alloc((size_t)N64 * 64 * 4);

  auto B = [](int n) { return dim3((unsigned)((n + 255) / 256)); };
  const size_t shB1 = (size_t)27 * 1 * 512 * 2;  // KB=1 LDS stage bytes
  const size_t shB2 = (size_t)27 * 2 * 512 * 2;  // KB=2

  // ---- scale 0.5 (32^3), 2 iterations ----
  vr_downsample_kernel<<<B(N32), 256, 0, stream>>>(image, moving, mask, img32, mov32, msk32);
  vr_fill_kernel<<<B(3 * N32), 256, 0, stream>>>(vf32, 0.f, 3 * N32);
  for (int it = 0; it < 2; ++it) {
    vr_warp_kernel<<<B(N32), 256, 0, stream>>>(mov32, vf32, warped, 32, 32, 32);
    vr_demons_update_kernel<<<B(N32), 256, 0, stream>>>(warped, img32, msk32, vf32, 32, 32, 32);
    vr_smooth_axis_kernel<<<B(3 * N32), 256, 0, stream>>>(vf32, tmpA, 32, 32, 32, 0);
    vr_smooth_axis_kernel<<<B(3 * N32), 256, 0, stream>>>(tmpA, tmpB, 32, 32, 32, 1);
    vr_smooth_axis_kernel<<<B(3 * N32), 256, 0, stream>>>(tmpB, vf32, 32, 32, 32, 2);
  }
  vr_upsample_vf_kernel<<<B(3 * N64), 256, 0, stream>>>(vf32, vf64);

  // ---- scale 1.0 (64^3), 2 iterations ----
  for (int it = 0; it < 2; ++it) {
    vr_warp_kernel<<<B(N64), 256, 0, stream>>>(moving, vf64, warped, 64, 64, 64);
    vr_demons_update_kernel<<<B(N64), 256, 0, stream>>>(warped, image, mask, vf64, 64, 64, 64);
    vr_smooth_axis_kernel<<<B(3 * N64), 256, 0, stream>>>(vf64, tmpA, 64, 64, 64, 0);
    vr_smooth_axis_kernel<<<B(3 * N64), 256, 0, stream>>>(tmpA, tmpB, 64, 64, 64, 1);
    vr_smooth_axis_kernel<<<B(3 * N64), 256, 0, stream>>>(tmpB, vf64, 64, 64, 64, 2);
  }

  // ---- CNN: stacked -> conv1 -> conv2 -> conv3 -> conv4 ----
  vr_build_act0_kernel<<<B(N64 * 32), 256, 0, stream>>>(vf64, image, mask, warped, actA);

  // conv1: cin=5(pad32), cout=32 (Ntiles=2, KB=1)
  vr_wprep_kernel<<<B(2 * 27 * 1 * 512), 256, 0, stream>>>(w1, wfrag, 32, 5, 2, 1);
  vr_conv3d_wmma_kernel<<<dim3(512, 2), 256, shB1, stream>>>(actA, wfrag, convout, 32, 32, 1);
  vr_fill_kernel<<<1, 128, 0, stream>>>(sums, 0.f, 128);
  vr_inorm_reduce_kernel<<<dim3(512), 256, 0, stream>>>(convout, sums, 32, N64);
  vr_inorm_mish_pack_kernel<<<B(N64 * 32), 256, 0, stream>>>(convout, sums, actB, 32, 32, 32, N64);

  // conv2: cin=32, cout=64 (Ntiles=4, KB=1)
  vr_wprep_kernel<<<B(4 * 27 * 1 * 512), 256, 0, stream>>>(w2, wfrag, 64, 32, 4, 1);
  vr_conv3d_wmma_kernel<<<dim3(512, 4), 256, shB1, stream>>>(actB, wfrag, convout, 32, 64, 1);
  vr_fill_kernel<<<1, 128, 0, stream>>>(sums, 0.f, 128);
  vr_inorm_reduce_kernel<<<dim3(512), 256, 0, stream>>>(convout, sums, 64, N64);
  vr_inorm_mish_pack_kernel<<<B(N64 * 64), 256, 0, stream>>>(convout, sums, actA, 64, 64, 64, N64);

  // conv3: cin=64, cout=32 (Ntiles=2, KB=2)
  vr_wprep_kernel<<<B(2 * 27 * 2 * 512), 256, 0, stream>>>(w3, wfrag, 32, 64, 2, 2);
  vr_conv3d_wmma_kernel<<<dim3(512, 2), 256, shB2, stream>>>(actA, wfrag, convout, 64, 32, 2);
  vr_fill_kernel<<<1, 128, 0, stream>>>(sums, 0.f, 128);
  vr_inorm_reduce_kernel<<<dim3(512), 256, 0, stream>>>(convout, sums, 32, N64);
  vr_inorm_mish_pack_kernel<<<B(N64 * 32), 256, 0, stream>>>(convout, sums, actB, 32, 32, 32, N64);

  // conv4: cin=32, cout=3(pad16) (Ntiles=1, KB=1), bias added in add_corr
  vr_wprep_kernel<<<B(1 * 27 * 1 * 512), 256, 0, stream>>>(w4, wfrag, 3, 32, 1, 1);
  vr_conv3d_wmma_kernel<<<dim3(512, 1), 256, shB1, stream>>>(actB, wfrag, convout, 32, 16, 1);

  // cvf = smooth(vf + correction + bias)
  vr_add_corr_kernel<<<B(3 * N64), 256, 0, stream>>>(vf64, convout, b4, tmpA);
  vr_smooth_axis_kernel<<<B(3 * N64), 256, 0, stream>>>(tmpA, tmpB, 64, 64, 64, 0);
  vr_smooth_axis_kernel<<<B(3 * N64), 256, 0, stream>>>(tmpB, tmpA, 64, 64, 64, 1);
  vr_smooth_axis_kernel<<<B(3 * N64), 256, 0, stream>>>(tmpA, cvf, 64, 64, 64, 2);

  // outputs: warp(moving,cvf), warp(moving,vf), cvf, vf
  vr_warp_kernel<<<B(N64), 256, 0, stream>>>(moving, cvf, out, 64, 64, 64);
  vr_warp_kernel<<<B(N64), 256, 0, stream>>>(moving, vf64, out + N64, 64, 64, 64);
  vr_copy_kernel<<<B(3 * N64), 256, 0, stream>>>(cvf, out + 2 * N64, 3 * N64);
  vr_copy_kernel<<<B(3 * N64), 256, 0, stream>>>(vf64, out + 5 * N64, 3 * N64);
}
